// LinearAttention_40226663694734
// MI455X (gfx1250) — matrix operand order repaired and checked by reading, not compiled, not run on hardware
//
#include <hip/hip_runtime.h>

// ---------------------------------------------------------------------------
// LinearAttention for MI455X (gfx1250): bf16 WMMA + Tensor Data Mover pipeline
//   qkv = x @ [Wq|Wk|Wv]  (TDM double-buffered WMMA GEMM, f32 acc -> bf16)
//   q   = softmax_d(q)*s  (wave-level reduction, d=64)
//   k   = softmax_n(k)    (block-level reduction, n=4096)
//   ctx = k @ v^T         (per-(b,h) WMMA, K=4096)
//   att = ctx^T @ q       (per-(b,h) WMMA, K=64)
//   out = att @ Wo + bo   (TDM double-buffered WMMA GEMM, f32 out + bias)
// B operands are pre-transposed in global memory so every tile is a plain 2D
// row-major tile -> ideal for tensor_load_to_lds (with LDS row padding via D#).
// ---------------------------------------------------------------------------

typedef __bf16 bf16_t;
typedef __attribute__((ext_vector_type(4)))  __bf16 v4bf;
typedef __attribute__((ext_vector_type(8)))  __bf16 v8bf;
typedef __attribute__((ext_vector_type(16))) __bf16 v16bf;
typedef __attribute__((ext_vector_type(8)))  float  v8f;
typedef __attribute__((ext_vector_type(4)))  unsigned int v4u;
typedef __attribute__((ext_vector_type(4)))  int v4i;
typedef __attribute__((ext_vector_type(8)))  int v8i;

#define NB      4
#define NSEQ    4096
#define QDIM    1024
#define HEADS   8
#define DHEAD   64
#define HID     512
#define QKVW    1536          // 3*HID columns of fused qkv GEMM
#define SCALE_Q 0.125f        // DIM_HEAD ** -0.5

__device__ __forceinline__ v16bf make_frag(v8bf lo, v8bf hi) {
  v16bf r;
#pragma unroll
  for (int i = 0; i < 8; ++i) { r[i] = lo[i]; r[i + 8] = hi[i]; }
  return r;
}

__device__ __forceinline__ void store_out(float* C, size_t idx, float v)  { C[idx] = v; }
__device__ __forceinline__ void store_out(bf16_t* C, size_t idx, float v) { C[idx] = (bf16_t)v; }

// ---------------------------------------------------------------------------
// Tensor Data Mover: 2D tile load global->LDS (D# per CDNA5 ISA ch.8).
// data_size = 2 bytes. LDS padding: +4 DWORDs every 16 DWORDs stored, i.e. a
// 32-elem (64B) tile row lands on an 80B (40-elem) LDS pitch -> matches the
// bank-conflict-free fragment layout used by the compute phase.
//   row_len   : tensor dim0 length == row stride (elements)
//   n_rows    : tensor dim1 length (rows)
//   tile_cols : tile dim0 (contiguous elements, here 32)
//   tile_rows : tile dim1 (rows)
// ---------------------------------------------------------------------------
__device__ __forceinline__ void tdm_load_tile_2d(unsigned lds_off, const bf16_t* gptr,
                                                 unsigned row_len, unsigned n_rows,
                                                 unsigned tile_cols, unsigned tile_rows) {
  unsigned long long ga = (unsigned long long)(uintptr_t)gptr;
  v4u g0;
  g0[0] = 1u;                                               // count=1, user desc
  g0[1] = lds_off;                                          // LDS byte address
  g0[2] = (unsigned)ga;                                     // global_addr[31:0]
  g0[3] = (unsigned)((ga >> 32) & 0x01ffffffu) | (2u << 30); // ga[56:32] | type=2
  v8i g1;
  g1[0] = (int)((1u << 16) |       // data_size = 1 -> 2 bytes
                (1u << 20) |       // pad_enable
                (3u << 22) |       // pad_interval = 3 -> every 16 DWORDs
                (3u << 25));       // pad_amount   = 3 -> 4 DWORDs
  g1[1] = (int)((row_len & 0xffffu) << 16);                     // tensor_dim0[15:0]
  g1[2] = (int)((row_len >> 16) | ((n_rows & 0xffffu) << 16));  // dim0 hi | dim1 lo
  g1[3] = (int)((n_rows >> 16) | (tile_cols << 16));            // dim1 hi | tile_dim0
  g1[4] = (int)tile_rows;                                       // tile_dim1 (tile_dim2=0)
  g1[5] = (int)row_len;                                         // tensor_dim0_stride lo
  g1[6] = 0;                                                    // stride0 hi | stride1 lo
  g1[7] = 0;
  v4i z4 = {0, 0, 0, 0};
#if defined(__clang_major__) && (__clang_major__ >= 23)
  v8i z8 = {0, 0, 0, 0, 0, 0, 0, 0};
  __builtin_amdgcn_tensor_load_to_lds(g0, g1, z4, z4, z8, 0);
#else
  __builtin_amdgcn_tensor_load_to_lds(g0, g1, z4, z4, 0);
#endif
}

// ---------------------------------------------------------------------------
// Converters / packers (B operands stored TRANSPOSED: [N][K] row-major)
// ---------------------------------------------------------------------------
__global__ void cvt_f32_bf16_v4(const float* __restrict__ in, bf16_t* __restrict__ out, int n4) {
  for (int i = blockIdx.x * blockDim.x + threadIdx.x; i < n4; i += gridDim.x * blockDim.x) {
    const float4 f = ((const float4*)in)[i];
    v4bf o; o[0] = (bf16_t)f.x; o[1] = (bf16_t)f.y; o[2] = (bf16_t)f.z; o[3] = (bf16_t)f.w;
    ((v4bf*)out)[i] = o;
  }
}

__global__ void pack_wqkv_t(const float* __restrict__ Wq, const float* __restrict__ Wk,
                            const float* __restrict__ Wv, bf16_t* __restrict__ outT) {
  // outT[n][k], n in [0,1536): 0..511 <- Wq, 512..1023 <- Wk, 1024..1535 <- Wv
  for (int i = blockIdx.x * blockDim.x + threadIdx.x; i < QKVW * QDIM;
       i += gridDim.x * blockDim.x) {
    const int n = i >> 10;          // / QDIM
    const int k = i & (QDIM - 1);
    float v;
    if (n < 512)       v = Wq[(size_t)k * HID + n];
    else if (n < 1024) v = Wk[(size_t)k * HID + (n - 512)];
    else               v = Wv[(size_t)k * HID + (n - 1024)];
    outT[i] = (bf16_t)v;
  }
}

__global__ void cvt_transpose_bf16(const float* __restrict__ in, bf16_t* __restrict__ outT,
                                   int rows, int cols) {
  // outT[c][r] = in[r][c]
  for (int i = blockIdx.x * blockDim.x + threadIdx.x; i < rows * cols;
       i += gridDim.x * blockDim.x) {
    const int c = i / rows;
    const int r = i - c * rows;
    outT[i] = (bf16_t)in[(size_t)r * cols + c];
  }
}

// ---------------------------------------------------------------------------
// TDM double-buffered WMMA GEMM: C[M,N] = A[M,K] @ Bt[N,K]^T (+bias)
// Block tile 128x64, 8 waves (4x2), wave tile 32x32 (2x2 WMMA), K-step 32.
// Wave 0 issues the tile DMAs and gates them with s_wait_tensorcnt; the
// workgroup barrier publishes the tile to all waves.
// ---------------------------------------------------------------------------
template <typename OutT>
__global__ __launch_bounds__(256) void gemm_bf16_wmma_tdm(
    const bf16_t* __restrict__ A, const bf16_t* __restrict__ Bt,
    const float* __restrict__ bias, OutT* __restrict__ C,
    int M, int N, int K) {
  __shared__ bf16_t As[2][128][40];   // [buf][m][k], 80B pitch (TDM-padded)
  __shared__ bf16_t Bs[2][64][40];    // [buf][n][k]

  const int tid  = threadIdx.x;
  const int lane = tid & 31;
  const int wave = tid >> 5;
  const int wm = wave >> 1, wn = wave & 1;
  const int blockM = blockIdx.y * 128;
  const int blockN = blockIdx.x * 64;

  const unsigned ldsA[2] = { (unsigned)(uintptr_t)&As[0][0][0],
                             (unsigned)(uintptr_t)&As[1][0][0] };
  const unsigned ldsB[2] = { (unsigned)(uintptr_t)&Bs[0][0][0],
                             (unsigned)(uintptr_t)&Bs[1][0][0] };
  const bf16_t* aBase = A  + (size_t)blockM * K;
  const bf16_t* bBase = Bt + (size_t)blockN * K;

  v8f acc[2][2] = {};

  const int nIter = K >> 5;
  if (tid < 32) {  // wave 0: prologue DMA into buffer 0
    tdm_load_tile_2d(ldsA[0], aBase, (unsigned)K, (unsigned)M, 32u, 128u);
    tdm_load_tile_2d(ldsB[0], bBase, (unsigned)K, (unsigned)N, 32u, 64u);
  }

  for (int it = 0; it < nIter; ++it) {
    const int cur = it & 1;
    if (tid < 32) {
      if (it + 1 < nIter) {
        const int k1 = (it + 1) << 5;
        tdm_load_tile_2d(ldsA[cur ^ 1], aBase + k1, (unsigned)K, (unsigned)M, 32u, 128u);
        tdm_load_tile_2d(ldsB[cur ^ 1], bBase + k1, (unsigned)K, (unsigned)N, 32u, 64u);
        __builtin_amdgcn_s_wait_tensorcnt(2);   // current tile landed; next in flight
      } else {
        __builtin_amdgcn_s_wait_tensorcnt(0);
      }
    }
    __syncthreads();   // publish current tile to all waves

    const int c0 = (lane >> 4) * 16;   // A-frag K chunk:  c0..c0+15
    const int kb = (lane >> 4) * 16;   // B-frag K chunk:  kb..kb+15
    v16bf af[2], bfr[2];
#pragma unroll
    for (int ti = 0; ti < 2; ++ti) {
      const int r = wm * 32 + ti * 16 + (lane & 15);
      af[ti] = make_frag(*(const v8bf*)&As[cur][r][c0], *(const v8bf*)&As[cur][r][c0 + 8]);
    }
#pragma unroll
    for (int tj = 0; tj < 2; ++tj) {
      const int n = wn * 32 + tj * 16 + (lane & 15);
      bfr[tj] = make_frag(*(const v8bf*)&Bs[cur][n][kb], *(const v8bf*)&Bs[cur][n][kb + 8]);
    }
#pragma unroll
    for (int ti = 0; ti < 2; ++ti)
#pragma unroll
      for (int tj = 0; tj < 2; ++tj)
        acc[ti][tj] = __builtin_amdgcn_wmma_f32_16x16x32_bf16(
            false, af[ti], false, bfr[tj], (short)0, acc[ti][tj], false, false);

    __syncthreads();   // all reads of buf[cur] done before it is re-filled
  }

#pragma unroll
  for (int ti = 0; ti < 2; ++ti)
#pragma unroll
    for (int tj = 0; tj < 2; ++tj) {
      const int col = blockN + wn * 32 + tj * 16 + (lane & 15);
      const float bvl = bias ? bias[col] : 0.0f;
#pragma unroll
      for (int i = 0; i < 8; ++i) {
        const int row = blockM + wm * 32 + ti * 16 + ((lane >> 4) ? (8 + i) : i);
        store_out(C, (size_t)row * N + col, acc[ti][tj][i] + bvl);
      }
    }
}

// ---------------------------------------------------------------------------
// softmax over d (64 contiguous elems) for q; writes q_sm[bh][d][n] bf16 *scale
// ---------------------------------------------------------------------------
__global__ __launch_bounds__(256) void softmax_q_kernel(const bf16_t* __restrict__ qkv,
                                                        bf16_t* __restrict__ qsm) {
  const int seg  = blockIdx.x * 8 + (threadIdx.x >> 5);   // 0 .. B*N*H-1
  const int lane = threadIdx.x & 31;
  const int b = seg >> 15;
  const int n = (seg >> 3) & 4095;
  const int h = seg & 7;
  const bf16_t* p = qkv + (size_t)(b * NSEQ + n) * QKVW + h * DHEAD + lane * 2;
  float v0 = (float)p[0], v1 = (float)p[1];
  float m = fmaxf(v0, v1);
#pragma unroll
  for (int off = 16; off; off >>= 1) m = fmaxf(m, __shfl_xor(m, off, 32));
  float e0 = __expf(v0 - m), e1 = __expf(v1 - m);
  float s = e0 + e1;
#pragma unroll
  for (int off = 16; off; off >>= 1) s += __shfl_xor(s, off, 32);
  const float inv = SCALE_Q / s;
  const size_t base = ((size_t)(b * HEADS + h) * DHEAD + lane * 2) * NSEQ + n;
  qsm[base]        = (bf16_t)(e0 * inv);
  qsm[base + NSEQ] = (bf16_t)(e1 * inv);
}

// ---------------------------------------------------------------------------
// softmax over n (4096, strided) for k; writes k_sm[bh][d][n] bf16
// ---------------------------------------------------------------------------
__global__ __launch_bounds__(256) void softmax_k_kernel(const bf16_t* __restrict__ qkv,
                                                        bf16_t* __restrict__ ksm) {
  __shared__ float red[8];
  const int col = blockIdx.x;                 // == (b*8+h)*64 + d
  const int b = col >> 9, h = (col >> 6) & 7, d = col & 63;
  const int tid = threadIdx.x;
  const size_t cbase = (size_t)512 + h * DHEAD + d;

  float vals[16];
  float m = -3.0e38f;
#pragma unroll
  for (int j = 0; j < 16; ++j) {
    const int n = tid + j * 256;
    vals[j] = (float)qkv[(size_t)(b * NSEQ + n) * QKVW + cbase];
    m = fmaxf(m, vals[j]);
  }
#pragma unroll
  for (int off = 16; off; off >>= 1) m = fmaxf(m, __shfl_xor(m, off, 32));
  if ((tid & 31) == 0) red[tid >> 5] = m;
  __syncthreads();
  float bm = red[0];
#pragma unroll
  for (int i = 1; i < 8; ++i) bm = fmaxf(bm, red[i]);
  __syncthreads();

  float s = 0.f;
#pragma unroll
  for (int j = 0; j < 16; ++j) { vals[j] = __expf(vals[j] - bm); s += vals[j]; }
#pragma unroll
  for (int off = 16; off; off >>= 1) s += __shfl_xor(s, off, 32);
  if ((tid & 31) == 0) red[tid >> 5] = s;
  __syncthreads();
  float bs = 0.f;
#pragma unroll
  for (int i = 0; i < 8; ++i) bs += red[i];
  const float inv = 1.0f / bs;

  const size_t obase = (size_t)col * NSEQ;
#pragma unroll
  for (int j = 0; j < 16; ++j) {
    const int n = tid + j * 256;
    ksm[obase + n] = (bf16_t)(vals[j] * inv);
  }
}

// ---------------------------------------------------------------------------
// context[d,e] = sum_n k_sm[d,n] * v[n,e]  per (b,h); stores ctx^T[e,d] bf16.
// ---------------------------------------------------------------------------
__global__ __launch_bounds__(512) void ctx_kernel(const bf16_t* __restrict__ ksm,
                                                  const bf16_t* __restrict__ qkv,
                                                  bf16_t* __restrict__ ctxT) {
  __shared__ bf16_t Ak[64][40];   // [d][k]
  __shared__ bf16_t Vt[64][40];   // transposed: [e][k]
  const int bh = blockIdx.x;
  const int b = bh >> 3, h = bh & 7;
  const int tid  = threadIdx.x;
  const int lane = tid & 31;
  const int wave = tid >> 5;
  const int mi = wave >> 2, ni = wave & 3;

  v8f acc = {};
  const int ad = tid >> 3, ac = (tid & 7) * 4;
  const int vr = tid >> 4, ve = (tid & 15) * 4;

  for (int k0 = 0; k0 < NSEQ; k0 += 32) {
    v4bf av = *(const v4bf*)(ksm + ((size_t)bh * DHEAD + ad) * NSEQ + k0 + ac);
    v4bf vv = *(const v4bf*)(qkv + ((size_t)(b * NSEQ) + k0 + vr) * QKVW + 1024 + h * DHEAD + ve);
    __syncthreads();
    *(v4bf*)&Ak[ad][ac] = av;
#pragma unroll
    for (int j = 0; j < 4; ++j) Vt[ve + j][vr] = vv[j];
    __syncthreads();

    const int nloc = ni * 16 + (lane & 15);
    const int kb   = (lane >> 4) * 16;
    v16bf bfrag = make_frag(*(const v8bf*)&Vt[nloc][kb], *(const v8bf*)&Vt[nloc][kb + 8]);
    const int row = mi * 16 + (lane & 15);
    const int c0  = (lane >> 4) * 16;
    v16bf afrag = make_frag(*(const v8bf*)&Ak[row][c0], *(const v8bf*)&Ak[row][c0 + 8]);
    acc = __builtin_amdgcn_wmma_f32_16x16x32_bf16(false, afrag, false, bfrag,
                                                  (short)0, acc, false, false);
  }

  const int ne = ni * 16 + (lane & 15);
#pragma unroll
  for (int i = 0; i < 8; ++i) {
    const int md = mi * 16 + ((lane >> 4) ? (8 + i) : i);
    ctxT[((size_t)bh * DHEAD + ne) * DHEAD + md] = (bf16_t)acc[i];
  }
}

// ---------------------------------------------------------------------------
// att[e,n] = sum_d ctxT[e,d] * q_sm[d,n]  per (b,h); writes attn[(b,n)][h*64+e]
// ---------------------------------------------------------------------------
__global__ __launch_bounds__(256) void attn_out_kernel(const bf16_t* __restrict__ ctxT,
                                                       const bf16_t* __restrict__ qsm,
                                                       bf16_t* __restrict__ attn) {
  __shared__ bf16_t Act[64][72];   // [e][d]
  __shared__ bf16_t Qt[128][72];   // transposed: [n][d]
  const int n0 = blockIdx.x * 128;
  const int bh = blockIdx.y;
  const int b = bh >> 3, h = bh & 7;
  const int tid  = threadIdx.x;
  const int lane = tid & 31;
  const int wave = tid >> 5;

  {
    const int e = tid >> 2, dseg = (tid & 3) * 16;
    const v8bf* src = (const v8bf*)(ctxT + ((size_t)bh * DHEAD + e) * DHEAD + dseg);
    *(v8bf*)&Act[e][dseg]     = src[0];
    *(v8bf*)&Act[e][dseg + 8] = src[1];
  }
  {
    const int d = tid >> 2, nseg = (tid & 3) * 32;
    const bf16_t* src = qsm + ((size_t)bh * DHEAD + d) * NSEQ + n0 + nseg;
#pragma unroll
    for (int j = 0; j < 32; ++j) Qt[nseg + j][d] = src[j];
  }
  __syncthreads();

  v8f acc[4] = {};
#pragma unroll
  for (int d0 = 0; d0 < 64; d0 += 32) {
    const int nloc = wave * 16 + (lane & 15);
    const int kk   = d0 + (lane >> 4) * 16;
    v16bf bfrag = make_frag(*(const v8bf*)&Qt[nloc][kk], *(const v8bf*)&Qt[nloc][kk + 8]);
#pragma unroll
    for (int mi = 0; mi < 4; ++mi) {
      const int e  = mi * 16 + (lane & 15);
      const int c0 = d0 + (lane >> 4) * 16;
      v16bf afrag = make_frag(*(const v8bf*)&Act[e][c0], *(const v8bf*)&Act[e][c0 + 8]);
      acc[mi] = __builtin_amdgcn_wmma_f32_16x16x32_bf16(false, afrag, false, bfrag,
                                                        (short)0, acc[mi], false, false);
    }
  }

  const int n = n0 + wave * 16 + (lane & 15);
#pragma unroll
  for (int mi = 0; mi < 4; ++mi)
#pragma unroll
    for (int i = 0; i < 8; ++i) {
      const int e = mi * 16 + ((lane >> 4) ? (8 + i) : i);
      attn[((size_t)(b * NSEQ) + n) * HID + h * DHEAD + e] = (bf16_t)acc[mi][i];
    }
}

// ---------------------------------------------------------------------------
// Host launcher
// ---------------------------------------------------------------------------
extern "C" void kernel_launch(void* const* d_in, const int* in_sizes, int n_in,
                              void* d_out, int out_size, void* d_ws, size_t ws_size,
                              hipStream_t stream) {
  (void)in_sizes; (void)n_in; (void)out_size; (void)ws_size;
  const float* x  = (const float*)d_in[0];
  const float* Wq = (const float*)d_in[1];
  const float* Wk = (const float*)d_in[2];
  const float* Wv = (const float*)d_in[3];
  const float* Wo = (const float*)d_in[4];
  const float* bo = (const float*)d_in[5];
  float* out = (float*)d_out;

  bf16_t* p = (bf16_t*)d_ws;
  bf16_t* xb    = p; p += (size_t)NB * NSEQ * QDIM;        // 16384x1024
  bf16_t* wqkvT = p; p += (size_t)QKVW * QDIM;             // 1536x1024 (transposed)
  bf16_t* woT   = p; p += (size_t)QDIM * HID;              // 1024x512  (transposed)
  bf16_t* qkv   = p; p += (size_t)NB * NSEQ * QKVW;        // 16384x1536
  bf16_t* qsm   = p; p += (size_t)NB * HEADS * DHEAD * NSEQ;
  bf16_t* ksm   = p; p += (size_t)NB * HEADS * DHEAD * NSEQ;
  bf16_t* ctxT  = p; p += (size_t)NB * HEADS * DHEAD * DHEAD;
  bf16_t* attn  = p;                                        // 16384x512

  cvt_f32_bf16_v4<<<4096, 256, 0, stream>>>(x, xb, (NB * NSEQ * QDIM) / 4);
  pack_wqkv_t<<<2048, 256, 0, stream>>>(Wq, Wk, Wv, wqkvT);
  cvt_transpose_bf16<<<1024, 256, 0, stream>>>(Wo, woT, HID, QDIM);

  // qkv = x @ Wqkv : M=16384, N=1536, K=1024
  gemm_bf16_wmma_tdm<bf16_t><<<dim3(QKVW / 64, (NB * NSEQ) / 128), 256, 0, stream>>>(
      xb, wqkvT, nullptr, qkv, NB * NSEQ, QKVW, QDIM);

  softmax_q_kernel<<<(NB * NSEQ * HEADS) / 8, 256, 0, stream>>>(qkv, qsm);
  softmax_k_kernel<<<NB * HEADS * DHEAD, 256, 0, stream>>>(qkv, ksm);

  ctx_kernel<<<NB * HEADS, 512, 0, stream>>>(ksm, qkv, ctxT);
  attn_out_kernel<<<dim3(NSEQ / 128, NB * HEADS), 256, 0, stream>>>(ctxT, qsm, attn);

  // out = attn @ Wo + bo : M=16384, N=1024, K=512
  gemm_bf16_wmma_tdm<float><<<dim3(QDIM / 64, (NB * NSEQ) / 128), 256, 0, stream>>>(
      attn, woT, bo, out, NB * NSEQ, QDIM, HID);
}
